// MinGRU_65103114272831
// MI455X (gfx1250) — matrix operand order repaired and checked by reading, not compiled, not run on hardware
//
#include <hip/hip_runtime.h>
#include <stdint.h>
#include <stddef.h>

// ---------------------------------------------------------------------------
// MinGRU forward for MI455X (gfx1250, wave32, WMMA)
//   z  = sigmoid(x @ Wz^T + bz)        (M=16384, N=1024, K=1024)
//   ht = x @ Wh^T + bh
//   a  = 1 - z + 1e-8 ;  b = z * ht
//   h_t = a_t * h_{t-1} + b_t          (chunked parallel scan)
//
// GEMM: split-precision bf16 (hi*hi + hi*lo + lo*hi) on
// v_wmma_f32_16x16x32_bf16. x/W hi-lo splits are PRECOMPUTED (memory-bound
// pass) so the WMMA loop has no conversion VALU. The block's shared A slab
// (32 rows x full K, hi+lo = 128KB) is staged in LDS via CDNA5 async
// global->LDS loads and read back with ds_load_b128.
// ---------------------------------------------------------------------------

typedef __attribute__((ext_vector_type(16))) __bf16        bf16x16;
typedef __attribute__((ext_vector_type(8)))  float         f32x8;
typedef __attribute__((ext_vector_type(4)))  unsigned int  u32x4;

// Pointee type of the async-LDS builtins (GCC vector, per clang diagnostic:
// "__attribute__((__vector_size__(4 * sizeof(int)))) int __device__ *")
typedef int v4i __attribute__((vector_size(16)));
typedef __attribute__((address_space(1))) v4i* glob_v4i_p;   // __device__ (AS1)
typedef __attribute__((address_space(3))) v4i* lds_v4i_p;    // __shared__ (AS3)

static constexpr int BATCH = 4;
static constexpr int T     = 4096;
static constexpr int D     = 1024;
static constexpr int M     = BATCH * T;   // 16384 rows
static constexpr int K     = D;
static constexpr int N     = D;
static constexpr int CH    = 64;          // chunks per sequence
static constexpr int CL    = T / CH;      // 64 timesteps per chunk

// LDS slab: [2 arrays (hi,lo)][32 rows][K * 2 bytes] = 2*32*2048 = 128 KB
static constexpr int LDS_ROW_BYTES = K * 2;              // 2048
static constexpr int LDS_ARR_BYTES = 32 * LDS_ROW_BYTES; // 65536
static constexpr int LDS_BYTES     = 2 * LDS_ARR_BYTES;  // 131072

#if defined(__has_builtin)
#if __has_builtin(__builtin_amdgcn_global_load_async_to_lds_b128)
#define HAVE_ASYNC_LDS 1
#endif
#endif

__device__ __forceinline__ void wait_async_lds() {
#if defined(__has_builtin)
#if __has_builtin(__builtin_amdgcn_s_wait_asynccnt)
    __builtin_amdgcn_s_wait_asynccnt(0);
#else
    asm volatile("s_wait_asynccnt 0" ::: "memory");
#endif
#else
    asm volatile("s_wait_asynccnt 0" ::: "memory");
#endif
}

__device__ __forceinline__ uint16_t f32_to_bf16_rne(float f) {
    uint32_t u = __builtin_bit_cast(uint32_t, f);
    u += 0x7FFFu + ((u >> 16) & 1u);
    return (uint16_t)(u >> 16);
}
__device__ __forceinline__ float bf16_to_f32(uint16_t h) {
    uint32_t u = ((uint32_t)h) << 16;
    return __builtin_bit_cast(float, u);
}
__device__ __forceinline__ bf16x16 load_bf16x16(const void* p) {
    union { bf16x16 v; u32x4 q[2]; } u;
    u.q[0] = *(const u32x4*)(p);
    u.q[1] = *(const u32x4*)((const char*)p + 16);
    return u.v;
}

// ----------------------- hi/lo bf16 split (x and W) ------------------------
__global__ __launch_bounds__(256) void split_kernel(
    const float* __restrict__ w, uint16_t* __restrict__ hi, uint16_t* __restrict__ lo)
{
    int i = blockIdx.x * 256 + threadIdx.x;
    float f = w[i];
    uint16_t h = f32_to_bf16_rne(f);
    hi[i] = h;
    lo[i] = f32_to_bf16_rne(f - bf16_to_f32(h));
}

// ------------------------ fused dual-GEMM + gating -------------------------
// 8 waves/block share one 32-row A slab (staged in LDS, full K, hi+lo).
// Each wave computes a 32M x 32N tile for BOTH weight matrices.
__global__ __launch_bounds__(256) void gemm_gate_kernel(
    const uint16_t* __restrict__ xh,  const uint16_t* __restrict__ xl,
    const uint16_t* __restrict__ wzh, const uint16_t* __restrict__ wzl,
    const uint16_t* __restrict__ whh, const uint16_t* __restrict__ whl,
    const float*    __restrict__ bz,  const float*    __restrict__ bh,
    float* __restrict__ a_out,   // == d_out, reused as scratch for `a`
    float* __restrict__ b_out)
{
    extern __shared__ __align__(16) char smem[];

    const int tid    = threadIdx.x;
    const int lane   = tid & 31;
    const int waveId = blockIdx.x * 8 + (tid >> 5);
    const int nTiles = N / 32;                 // 32; 8 waves/block => all
    const int mTile  = waveId / nTiles;        // waves in a block share mTile
    const int m0     = mTile * 32;
    const int n0     = (waveId % nTiles) * 32;
    const int l15    = lane & 15;
    const int hi16   = lane >> 4;

    // ---- stage A slab (rows m0..m0+31, all K, hi then lo) into LDS ----
    {
        const char* gsrc[2] = { (const char*)(xh + (size_t)m0 * K),
                                (const char*)(xl + (size_t)m0 * K) };
#pragma unroll
        for (int arr = 0; arr < 2; ++arr) {
            const char* g = gsrc[arr];
            char*       l = smem + arr * LDS_ARR_BYTES;
            // 65536 B per array = 4096 chunks of 16B; 256 threads -> 16 iters
            for (int cid = tid; cid < LDS_ARR_BYTES / 16; cid += 256) {
                int off = cid * 16;
#if defined(HAVE_ASYNC_LDS)
                __builtin_amdgcn_global_load_async_to_lds_b128(
                    (glob_v4i_p)(g + off), (lds_v4i_p)(l + off), 0, 0);
#else
                *(u32x4*)(l + off) = *(const u32x4*)(g + off);
#endif
            }
        }
#if defined(HAVE_ASYNC_LDS)
        wait_async_lds();
#endif
        __syncthreads();
    }

    f32x8 accz[2][2] = {};
    f32x8 acch[2][2] = {};

    for (int k0 = 0; k0 < K; k0 += 32) {
        // ---- A fragments from LDS ----
        // A k order used here (contiguous, not the ISA table): lanes0-15 hold K 0..15,
        // lanes16-31 hold K 16..31 (matches B); lane%16 selects row M.
        bf16x16 Ah[2], Al[2];
#pragma unroll
        for (int mi = 0; mi < 2; ++mi) {
            int rowOff = (mi * 16 + l15) * LDS_ROW_BYTES + (k0 + hi16 * 16) * 2;
            Ah[mi] = load_bf16x16(smem + rowOff);
            Al[mi] = load_bf16x16(smem + LDS_ARR_BYTES + rowOff);
        }

        // ---- B fragments from global (weights live in L2) ----
        // lane%16 = column n = row of W (contiguous K);
        // lanes0-15 hold K k0+0..15, lanes16-31 hold K k0+16..31.
        bf16x16 Bzh[2], Bzl[2], Bhh[2], Bhl[2];
#pragma unroll
        for (int ni = 0; ni < 2; ++ni) {
            size_t off = (size_t)(n0 + ni * 16 + l15) * K + k0 + hi16 * 16;
            Bzh[ni] = load_bf16x16(wzh + off);
            Bzl[ni] = load_bf16x16(wzl + off);
            Bhh[ni] = load_bf16x16(whh + off);
            Bhl[ni] = load_bf16x16(whl + off);
        }

        // ---- split-precision MACs: hi*hi + hi*lo + lo*hi ----
#pragma unroll
        for (int mi = 0; mi < 2; ++mi)
#pragma unroll
            for (int ni = 0; ni < 2; ++ni) {
                accz[mi][ni] = __builtin_amdgcn_wmma_f32_16x16x32_bf16(
                    false, Ah[mi], false, Bzh[ni], (short)0, accz[mi][ni], false, false);
                accz[mi][ni] = __builtin_amdgcn_wmma_f32_16x16x32_bf16(
                    false, Ah[mi], false, Bzl[ni], (short)0, accz[mi][ni], false, false);
                accz[mi][ni] = __builtin_amdgcn_wmma_f32_16x16x32_bf16(
                    false, Al[mi], false, Bzh[ni], (short)0, accz[mi][ni], false, false);
                acch[mi][ni] = __builtin_amdgcn_wmma_f32_16x16x32_bf16(
                    false, Ah[mi], false, Bhh[ni], (short)0, acch[mi][ni], false, false);
                acch[mi][ni] = __builtin_amdgcn_wmma_f32_16x16x32_bf16(
                    false, Ah[mi], false, Bhl[ni], (short)0, acch[mi][ni], false, false);
                acch[mi][ni] = __builtin_amdgcn_wmma_f32_16x16x32_bf16(
                    false, Al[mi], false, Bhh[ni], (short)0, acch[mi][ni], false, false);
            }
    }

    // ---- fused epilogue: bias + sigmoid + a/b for the scan ----
    // C/D layout: lanes0-15 col N=lane, VGPR r -> row M=r (lanes16-31: M=r+8)
#pragma unroll
    for (int mi = 0; mi < 2; ++mi)
#pragma unroll
        for (int ni = 0; ni < 2; ++ni) {
            const int e = n0 + ni * 16 + l15;
            const float bzv = bz[e];
            const float bhv = bh[e];
#pragma unroll
            for (int r = 0; r < 8; ++r) {
                const int m = m0 + mi * 16 + r + hi16 * 8;
                float z  = 1.0f / (1.0f + __expf(-(accz[mi][ni][r] + bzv)));
                float av = 1.0f - z + 1e-8f;
                float bv = z * (acch[mi][ni][r] + bhv);
                size_t idx = (size_t)m * D + e;
                a_out[idx] = av;
                b_out[idx] = bv;
            }
        }
}

// -------------------------- chunked linear scan ----------------------------
__global__ __launch_bounds__(256) void scan_pass1_kernel(
    const float* __restrict__ a, const float* __restrict__ b,
    float* __restrict__ Ac, float* __restrict__ Bc)
{
    int gid = blockIdx.x * 256 + threadIdx.x;      // B*D*CH threads
    int e  = gid % D;
    int c  = (gid / D) % CH;
    int bi = gid / (D * CH);
    size_t base = ((size_t)bi * T + (size_t)c * CL) * D + e;
    float A = 1.0f, Bv = 0.0f;
    for (int t = 0; t < CL; ++t) {
        size_t idx = base + (size_t)t * D;
        __builtin_prefetch(&a[idx + (size_t)8 * D], 0, 1);
        __builtin_prefetch(&b[idx + (size_t)8 * D], 0, 1);
        float av = a[idx];
        float bv = b[idx];
        Bv = av * Bv + bv;
        A *= av;
    }
    Ac[gid] = A;
    Bc[gid] = Bv;
}

__global__ __launch_bounds__(256) void scan_pass2_kernel(
    const float* __restrict__ Ac, const float* __restrict__ Bc,
    float* __restrict__ H0)
{
    int tid = blockIdx.x * 256 + threadIdx.x;      // B*D threads
    int e  = tid % D;
    int bi = tid / D;
    float h = 0.0f;
    for (int c = 0; c < CH; ++c) {
        size_t idx = ((size_t)bi * CH + c) * D + e;
        H0[idx] = h;
        h = Ac[idx] * h + Bc[idx];
    }
}

__global__ __launch_bounds__(256) void scan_pass3_kernel(
    const float* __restrict__ b, const float* __restrict__ H0,
    float* __restrict__ a_h)   // in: a, out: h  (same buffer = d_out)
{
    int gid = blockIdx.x * 256 + threadIdx.x;      // B*D*CH threads
    int e  = gid % D;
    int c  = (gid / D) % CH;
    int bi = gid / (D * CH);
    size_t base = ((size_t)bi * T + (size_t)c * CL) * D + e;
    float h = H0[gid];
    for (int t = 0; t < CL; ++t) {
        size_t idx = base + (size_t)t * D;
        __builtin_prefetch(&a_h[idx + (size_t)8 * D], 0, 1);
        __builtin_prefetch(&b[idx + (size_t)8 * D], 0, 1);
        h = a_h[idx] * h + b[idx];
        a_h[idx] = h;
    }
}

// ---------------------------------------------------------------------------
extern "C" void kernel_launch(void* const* d_in, const int* in_sizes, int n_in,
                              void* d_out, int out_size, void* d_ws, size_t ws_size,
                              hipStream_t stream) {
    (void)in_sizes; (void)n_in; (void)out_size; (void)ws_size;
    const float* x  = (const float*)d_in[0];
    const float* Wz = (const float*)d_in[1];
    const float* bz = (const float*)d_in[2];
    const float* Wh = (const float*)d_in[3];
    const float* bh = (const float*)d_in[4];
    float* a_h = (float*)d_out;                    // holds a, then final h

    char* ws = (char*)d_ws;
    size_t off = 0;
    float*    b_buf = (float*)(ws + off);    off += (size_t)M * D * sizeof(float);   // 64MB
    uint16_t* xh    = (uint16_t*)(ws + off); off += (size_t)M * K * sizeof(uint16_t); // 32MB
    uint16_t* xl    = (uint16_t*)(ws + off); off += (size_t)M * K * sizeof(uint16_t); // 32MB
    uint16_t* wzh   = (uint16_t*)(ws + off); off += (size_t)N * K * sizeof(uint16_t);
    uint16_t* wzl   = (uint16_t*)(ws + off); off += (size_t)N * K * sizeof(uint16_t);
    uint16_t* whh   = (uint16_t*)(ws + off); off += (size_t)N * K * sizeof(uint16_t);
    uint16_t* whl   = (uint16_t*)(ws + off); off += (size_t)N * K * sizeof(uint16_t);
    float*    Ac    = (float*)(ws + off);    off += (size_t)BATCH * D * CH * sizeof(float);
    float*    Bc    = (float*)(ws + off);    off += (size_t)BATCH * D * CH * sizeof(float);
    float*    H0    = (float*)(ws + off);    off += (size_t)BATCH * D * CH * sizeof(float);

    // 1) hi/lo bf16 splits (memory-bound; removes all conversion VALU
    //    from the WMMA loop and halves its read traffic)
    split_kernel<<<(M * K) / 256, 256, 0, stream>>>(x, xh, xl);
    split_kernel<<<(N * K) / 256, 256, 0, stream>>>(Wz, wzh, wzl);
    split_kernel<<<(N * K) / 256, 256, 0, stream>>>(Wh, whh, whl);

    // 2) fused dual GEMM + gating (16384 wave tiles, 8 waves/block,
    //    128KB dynamic LDS for the shared A slab)
    gemm_gate_kernel<<<(M / 32) * (N / 32) / 8, 256, LDS_BYTES, stream>>>(
        xh, xl, wzh, wzl, whh, whl, bz, bh, a_h, b_buf);

    // 3) chunked scan
    scan_pass1_kernel<<<(BATCH * D * CH) / 256, 256, 0, stream>>>(a_h, b_buf, Ac, Bc);
    scan_pass2_kernel<<<(BATCH * D) / 256, 256, 0, stream>>>(Ac, Bc, H0);
    scan_pass3_kernel<<<(BATCH * D * CH) / 256, 256, 0, stream>>>(b_buf, H0, a_h);
}
